// NatureCNN_63505386438864
// MI455X (gfx1250) — matrix operand order repaired and checked by reading, not compiled, not run on hardware
//
#include <hip/hip_runtime.h>
#include <hip/hip_bf16.h>

typedef __attribute__((ext_vector_type(16))) _Float16 v16h;
typedef __attribute__((ext_vector_type(8)))  _Float16 v8h;
typedef __attribute__((ext_vector_type(4)))  _Float16 v4h;
typedef __attribute__((ext_vector_type(2)))  _Float16 v2h;
typedef __attribute__((ext_vector_type(8)))  float    v8f;
typedef __attribute__((ext_vector_type(4)))  float    v4f;

union U16 {
  v16h v;
  v8h  h8[2];
  v4h  h4[4];
  v2h  h2[8];
  _Float16 e[16];
};

// D = A(16x32 f16) * B(32x16 f16) + C(16x16 f32)
__device__ __forceinline__ v8f wmma_f16(v16h a, v16h b, v8f c) {
  return __builtin_amdgcn_wmma_f32_16x16x32_f16(false, a, false, b, (short)0, c,
                                                false, false);
}

// ---- CDNA5 async global->LDS copy (VGPR-free staging, tracked by ASYNCcnt) ----
// Flat LDS pointer low 32 bits == LDS byte offset (ISA 10.2 aperture mapping).
__device__ __forceinline__ void async_ld_b128(void* lds_p, const void* gp) {
  unsigned off = (unsigned)(unsigned long long)lds_p;
  asm volatile("global_load_async_to_lds_b128 %0, %1, off"
               :: "v"(off), "v"(gp) : "memory");
}
__device__ __forceinline__ void async_wait0() {
  asm volatile("s_wait_asynccnt 0" ::: "memory");
}

// ---------------- weight prep: plain f32 -> f16 cast (weights stay [N][K]) ----
__global__ void cast_f16_kernel(_Float16* __restrict__ dst,
                                const float* __restrict__ src, int n8) {
  int i = blockIdx.x * blockDim.x + threadIdx.x;
  if (i < n8) {
    v4f a = *(const v4f*)&src[(size_t)i * 8];
    v4f b = *(const v4f*)&src[(size_t)i * 8 + 4];
    v8h o;
    o[0] = (_Float16)a.x; o[1] = (_Float16)a.y;
    o[2] = (_Float16)a.z; o[3] = (_Float16)a.w;
    o[4] = (_Float16)b.x; o[5] = (_Float16)b.y;
    o[6] = (_Float16)b.z; o[7] = (_Float16)b.w;
    *(v8h*)&dst[(size_t)i * 8] = o;
  }
}

// head16[32][256]: rows 0..17 actor, row 18 critic, rows 19..31 zero. headB[32].
__global__ void head_build_kernel(_Float16* __restrict__ head16,
                                  float* __restrict__ headB,
                                  const float* __restrict__ aw,
                                  const float* __restrict__ ab,
                                  const float* __restrict__ cw,
                                  const float* __restrict__ cb) {
  int idx = blockIdx.x * blockDim.x + threadIdx.x;
  if (idx < 32 * 256) {
    int n = idx >> 8, k = idx & 255;
    float v = 0.f;
    if (n < 18) v = aw[n * 256 + k];
    else if (n == 18) v = cw[k];
    head16[idx] = (_Float16)v;
  }
  if (idx < 32) {
    float v = 0.f;
    if (idx < 18) v = ab[idx];
    else if (idx == 18) v = cb[0];
    headB[idx] = v;
  }
}

// ---------------- conv1: 4x64x64 -> 32x15x15 (pad 16), k=8 s=4  (M=225,N=32,K=256)
__global__ __launch_bounds__(256) void conv1_kernel(
    const float* __restrict__ x, const _Float16* __restrict__ w16,
    const float* __restrict__ bias, _Float16* __restrict__ out) {
  __shared__ _Float16 simg[4 * 64 * 64];   // 32 KB
  __shared__ _Float16 sw[32 * 256];        // 16 KB, [n][k]
  const int img = blockIdx.x, tid = threadIdx.x;
  const float* xi = x + (size_t)img * 16384;
  // weights: async global->LDS (no conversion needed) ...
  for (int g = tid; g < 1024; g += 256)
    async_ld_b128(&sw[g * 8], &w16[g * 8]);
  // ... overlapped with f32->f16 image conversion through VGPRs
  for (int g = tid; g < 4096; g += 256) {
    v4f f = *(const v4f*)&xi[g * 4];
    v4h h;
    h[0] = (_Float16)(f.x * (1.0f / 255.0f));
    h[1] = (_Float16)(f.y * (1.0f / 255.0f));
    h[2] = (_Float16)(f.z * (1.0f / 255.0f));
    h[3] = (_Float16)(f.w * (1.0f / 255.0f));
    *(v4h*)&simg[g * 4] = h;
  }
  async_wait0();
  __syncthreads();
  const int wv = tid >> 5, lane = tid & 31, hi = lane >> 4, lm = lane & 15;
  for (int tt = wv; tt < 30; tt += 8) {     // 15 M-tiles x 2 N-tiles
    const int tm = tt >> 1, tn = tt & 1;
    const int p = tm * 16 + lm;
    const int pc = p < 225 ? p : 224;       // clamp: rows >=225 never stored
    const int yb = (pc / 15) * 4, xb = (pc % 15) * 4;
    const int nB = tn * 16 + lm;
    v8f acc = {};
    for (int k0 = 0; k0 < 256; k0 += 32) {
      U16 a, b;
      int kA = k0 + (hi << 3);              // run of 8: fixed (c,r), s=0..7
      {
        int c = kA >> 6, r = (kA >> 3) & 7;
        const _Float16* p0 = &simg[c * 4096 + (yb + r) * 64 + xb];
        a.h4[0] = *(const v4h*)p0;
        a.h4[1] = *(const v4h*)(p0 + 4);
      }
      {
        int k2 = kA + 16;
        int c = k2 >> 6, r = (k2 >> 3) & 7;
        const _Float16* p1 = &simg[c * 4096 + (yb + r) * 64 + xb];
        a.h4[2] = *(const v4h*)p1;
        a.h4[3] = *(const v4h*)(p1 + 4);
      }
      const _Float16* bp = &sw[nB * 256 + k0 + (hi << 3)];
      b.h8[0] = *(const v8h*)bp;
      b.h8[1] = *(const v8h*)(bp + 16);
      acc = wmma_f16(a.v, b.v, acc);
    }
#pragma unroll
    for (int i = 0; i < 8; ++i) {
      int m = tm * 16 + i + (hi << 3);
      if (m < 225) {
        float v = acc[i] + bias[nB];
        v = v > 0.f ? v : 0.f;
        // padded layout [ch][15][16]
        out[(size_t)img * 7680 + nB * 240 + (m / 15) * 16 + (m % 15)] = (_Float16)v;
      }
    }
  }
}

// ---------------- conv2: 32x15x15(pad16) -> 64x6x6, k=4 s=2  (M=36,N=64,K=512)
__global__ __launch_bounds__(256) void conv2_kernel(
    const _Float16* __restrict__ in, const _Float16* __restrict__ w16,
    const float* __restrict__ bias, _Float16* __restrict__ out) {
  __shared__ _Float16 sin_[32 * 240];      // 15 KB, [c][15][16]
  __shared__ _Float16 swc[64 * 128];       // 16 KB K-chunk, [n][kk]
  const int img = blockIdx.x, tid = threadIdx.x;
  const _Float16* ii = in + (size_t)img * 7680;
  for (int g = tid; g < 960; g += 256)     // async input staging
    async_ld_b128(&sin_[g * 8], &ii[g * 8]);
  const int wv = tid >> 5, lane = tid & 31, hi = lane >> 4, lm = lane & 15;
  v8f acc[2] = {{}, {}};
  for (int c0 = 0; c0 < 512; c0 += 128) {
    __syncthreads();
    for (int g = tid; g < 1024; g += 256) { // async [64][128] weight chunk
      int n = g >> 4, k8 = (g & 15) * 8;
      async_ld_b128(&swc[n * 128 + k8], &w16[n * 512 + c0 + k8]);
    }
    async_wait0();
    __syncthreads();
    int slot = 0;
    for (int tt = wv; tt < 12; tt += 8, ++slot) {  // 3 M-tiles x 4 N-tiles
      const int tm = tt >> 2, tn = tt & 3;
      const int p = tm * 16 + lm;
      const int pc = p < 36 ? p : 35;
      const int yb = (pc / 6) * 2, xb = (pc % 6) * 2;
      const int nB = tn * 16 + lm;
      for (int kk = 0; kk < 128; kk += 32) {
        U16 a, b;
        int kA = c0 + kk + (hi << 3);
#pragma unroll
        for (int j = 0; j < 4; ++j) {       // 4-runs: fixed (c,r), s=0..3
          int kb = kA + ((j & 1) << 2) + ((j >> 1) << 4);
          int c = kb >> 4, r = (kb >> 2) & 3;
          const _Float16* pp = &sin_[c * 240 + (yb + r) * 16 + xb];
          a.h2[2 * j]     = *(const v2h*)pp;
          a.h2[2 * j + 1] = *(const v2h*)(pp + 2);
        }
        const _Float16* bp = &swc[nB * 128 + kk + (hi << 3)];
        b.h8[0] = *(const v8h*)bp;
        b.h8[1] = *(const v8h*)(bp + 16);
        acc[slot] = wmma_f16(a.v, b.v, acc[slot]);
      }
    }
  }
  int slot = 0;
  for (int tt = wv; tt < 12; tt += 8, ++slot) {
    const int tm = tt >> 2, tn = tt & 3;
    const int nB = tn * 16 + lm;
#pragma unroll
    for (int i = 0; i < 8; ++i) {
      int m = tm * 16 + i + (hi << 3);
      if (m < 36) {
        float v = acc[slot][i] + bias[nB];
        v = v > 0.f ? v : 0.f;
        out[(size_t)img * 2304 + nB * 36 + m] = (_Float16)v;  // [ch][p]
      }
    }
  }
}

// ---------------- conv3: 64x6x6 -> 64x4x4, k=3 s=1  (M=16,N=64,K=576)
__global__ __launch_bounds__(128) void conv3_kernel(
    const _Float16* __restrict__ in, const _Float16* __restrict__ w16,
    const float* __restrict__ bias, _Float16* __restrict__ out) {
  __shared__ _Float16 sin_[64 * 36];       // 4.5 KB
  __shared__ _Float16 swc[64 * 192];       // 24 KB K-chunk, [n][kk]
  const int img = blockIdx.x, tid = threadIdx.x;
  const _Float16* ii = in + (size_t)img * 2304;
  for (int g = tid; g < 288; g += 128)     // async input staging
    async_ld_b128(&sin_[g * 8], &ii[g * 8]);
  const int wv = tid >> 5, lane = tid & 31, hi = lane >> 4, lm = lane & 15;
  const int tn = wv;                        // 4 waves, one N tile each
  const int oy = lm >> 2, ox = lm & 3;      // M=16, p = lm
  const int nB = tn * 16 + lm;
  v8f acc = {};
  for (int c0 = 0; c0 < 576; c0 += 192) {
    __syncthreads();
    for (int g = tid; g < 1536; g += 128) { // async weight chunk
      int n = g / 24, k8 = (g % 24) * 8;
      async_ld_b128(&swc[n * 192 + k8], &w16[n * 576 + c0 + k8]);
    }
    async_wait0();
    __syncthreads();
    for (int kk = 0; kk < 192; kk += 32) {
      U16 a, b;
#pragma unroll
      for (int e = 0; e < 16; ++e) {        // 3x3 taps: scalar gather, no masks
        int k = c0 + kk + ((e & 8) << 1) + (e & 7) + (hi << 3);
        int c = k / 9, rs = k % 9, r = rs / 3, s = rs % 3;
        a.e[e] = sin_[c * 36 + (oy + r) * 6 + (ox + s)];
      }
      const _Float16* bp = &swc[nB * 192 + kk + (hi << 3)];
      b.h8[0] = *(const v8h*)bp;
      b.h8[1] = *(const v8h*)(bp + 16);
      acc = wmma_f16(a.v, b.v, acc);
    }
  }
#pragma unroll
  for (int i = 0; i < 8; ++i) {
    int m = i + (hi << 3);
    float v = acc[i] + bias[nB];
    v = v > 0.f ? v : 0.f;
    out[(size_t)img * 1024 + nB * 16 + m] = (_Float16)v;  // flatten c*16+y*4+x
  }
}

// ---------------- generic GEMM: out = act(A[M,K] @ B[N,K]^T + bias) ------------
__global__ __launch_bounds__(256) void gemm_kernel(
    const _Float16* __restrict__ A, const _Float16* __restrict__ B,
    const float* __restrict__ bias, _Float16* outH, float* outF,
    int M, int N, int K, int ldo, int Nstore, int relu) {
  __shared__ _Float16 As[64 * 32];          // [m][k]
  __shared__ _Float16 BsT[64 * 32];         // [n][k]
  const int m0 = blockIdx.x * 64, n0 = blockIdx.y * 64;
  const int tid = threadIdx.x, wv = tid >> 5, lane = tid & 31;
  const int hi = lane >> 4, lm = lane & 15;
  const int ldr = tid >> 2, ld8 = (tid & 3) * 8;     // 64 rows x 4 v8h chunks
  const int arow = min(m0 + ldr, M - 1);
  const int brow = min(n0 + ldr, N - 1);
  v8f acc[2] = {{}, {}};
  for (int k0 = 0; k0 < K; k0 += 32) {
    __syncthreads();
    // async global->LDS staging: one b128 per lane per tile, no VGPR round-trip
    async_ld_b128(&As[ldr * 32 + ld8],  &A[(size_t)arow * K + k0 + ld8]);
    async_ld_b128(&BsT[ldr * 32 + ld8], &B[(size_t)brow * K + k0 + ld8]);
    if (k0 + 32 < K)  // gfx1250 global_prefetch_b8 for the next A K-slab
      __builtin_prefetch((const void*)(A + (size_t)arow * K + k0 + 32), 0, 1);
    async_wait0();
    __syncthreads();
#pragma unroll
    for (int s = 0; s < 2; ++s) {
      int st = wv + s * 8, sm = st >> 2, sn = st & 3;
      U16 a, b;
      const _Float16* ap = &As[(sm * 16 + lm) * 32 + (hi << 3)];
      a.h8[0] = *(const v8h*)ap;
      a.h8[1] = *(const v8h*)(ap + 16);
      const _Float16* bp = &BsT[(sn * 16 + lm) * 32 + (hi << 3)];
      b.h8[0] = *(const v8h*)bp;
      b.h8[1] = *(const v8h*)(bp + 16);
      acc[s] = wmma_f16(a.v, b.v, acc[s]);
    }
  }
#pragma unroll
  for (int s = 0; s < 2; ++s) {
    int st = wv + s * 8, sm = st >> 2, sn = st & 3;
#pragma unroll
    for (int i = 0; i < 8; ++i) {
      int m = m0 + sm * 16 + i + (hi << 3);
      int n = n0 + sn * 16 + lm;
      if (m < M && n < Nstore) {
        float v = acc[s][i] + bias[n];
        if (relu) v = v > 0.f ? v : 0.f;
        if (outH) outH[(size_t)m * ldo + n] = (_Float16)v;
        else      outF[(size_t)m * ldo + n] = v;
      }
    }
  }
}

// ---------------- persistent single-workgroup LSTM (T=128, N=32, H=256) --------
__global__ __launch_bounds__(1024) void lstm_kernel(
    const _Float16* __restrict__ feats, const _Float16* __restrict__ wi16,
    const _Float16* __restrict__ wh16, const float* __restrict__ bi,
    const float* __restrict__ bh, const float* __restrict__ done,
    const float* __restrict__ h0, const float* __restrict__ c0,
    float* __restrict__ gates, _Float16* __restrict__ outs) {
  __shared__ _Float16 hf[32 * 256];   // h state f16 (WMMA A operand)
  __shared__ float    cf[32 * 256];   // c state f32
  const int tid = threadIdx.x, wv = tid >> 5, lane = tid & 31;
  const int hi = lane >> 4, lm = lane & 15;
  for (int q = tid; q < 8192; q += 1024) {
    hf[q] = (_Float16)h0[q];
    cf[q] = c0[q];
  }
  __syncthreads();
  for (int t = 0; t < 128; ++t) {
    for (int q = tid; q < 8192; q += 1024) {     // episode-start reset
      float m = 1.0f - done[t * 32 + (q >> 8)];
      cf[q] *= m;
      hf[q] = (_Float16)((float)hf[q] * m);
    }
    __syncthreads();
    // gates[32][1024] = feats_t @ Wi^T + h @ Wh^T  (128 tiles / 32 waves)
    for (int tt = wv; tt < 128; tt += 32) {
      int tm = tt >> 6, tn = tt & 63;
      int mrow = tm * 16 + lm;                    // batch row
      int nn = tn * 16 + lm;                      // gate column (weight row)
      const _Float16* fr = feats + (size_t)(t * 32 + mrow) * 512;
      v8f acc = {};
      for (int k0 = 0; k0 < 512; k0 += 32) {      // input part
        U16 a, b;
        const _Float16* ap = fr + k0 + (hi << 3);
        a.h8[0] = *(const v8h*)ap;
        a.h8[1] = *(const v8h*)(ap + 16);
        const _Float16* bp = wi16 + (size_t)nn * 512 + k0 + (hi << 3);
        b.h8[0] = *(const v8h*)bp;
        b.h8[1] = *(const v8h*)(bp + 16);
        acc = wmma_f16(a.v, b.v, acc);
      }
      for (int k0 = 0; k0 < 256; k0 += 32) {      // hidden part
        U16 a, b;
        const _Float16* ap = &hf[mrow * 256 + k0 + (hi << 3)];
        a.h8[0] = *(const v8h*)ap;
        a.h8[1] = *(const v8h*)(ap + 16);
        const _Float16* bp = wh16 + (size_t)nn * 256 + k0 + (hi << 3);
        b.h8[0] = *(const v8h*)bp;
        b.h8[1] = *(const v8h*)(bp + 16);
        acc = wmma_f16(a.v, b.v, acc);
      }
#pragma unroll
      for (int i = 0; i < 8; ++i)
        gates[(tm * 16 + i + (hi << 3)) * 1024 + nn] = acc[i];
    }
    __threadfence();
    __syncthreads();
    for (int q = tid; q < 8192; q += 1024) {      // pointwise update
      int b = q >> 8, u = q & 255;
      float ig = gates[b * 1024 + u]       + bi[u]       + bh[u];
      float fg = gates[b * 1024 + 256 + u] + bi[256 + u] + bh[256 + u];
      float gg = gates[b * 1024 + 512 + u] + bi[512 + u] + bh[512 + u];
      float og = gates[b * 1024 + 768 + u] + bi[768 + u] + bh[768 + u];
      float is = 1.0f / (1.0f + expf(-ig));
      float fs = 1.0f / (1.0f + expf(-fg));
      float os = 1.0f / (1.0f + expf(-og));
      float c  = fs * cf[q] + is * tanhf(gg);
      float h  = os * tanhf(c);
      cf[q] = c;
      hf[q] = (_Float16)h;
      outs[(size_t)(t * 32 + b) * 256 + u] = (_Float16)h;
    }
    __syncthreads();
  }
}

extern "C" void kernel_launch(void* const* d_in, const int* in_sizes, int n_in,
                              void* d_out, int out_size, void* d_ws, size_t ws_size,
                              hipStream_t stream) {
  const float* x    = (const float*)d_in[0];
  const float* done = (const float*)d_in[1];
  const float* h0   = (const float*)d_in[2];
  const float* c0   = (const float*)d_in[3];
  const float* w1   = (const float*)d_in[4];
  const float* b1   = (const float*)d_in[5];
  const float* w2   = (const float*)d_in[6];
  const float* b2   = (const float*)d_in[7];
  const float* w3   = (const float*)d_in[8];
  const float* b3   = (const float*)d_in[9];
  const float* fw   = (const float*)d_in[10];
  const float* fb   = (const float*)d_in[11];
  const float* wi   = (const float*)d_in[12];
  const float* wh   = (const float*)d_in[13];
  const float* bi   = (const float*)d_in[14];
  const float* bh   = (const float*)d_in[15];
  const float* aw   = (const float*)d_in[16];
  const float* ab   = (const float*)d_in[17];
  const float* cw   = (const float*)d_in[18];
  const float* cb   = (const float*)d_in[19];

  char* p = (char*)d_ws;
  auto alloc = [&](size_t bytes) -> void* {
    void* r = (void*)p;
    p += (bytes + 255) & ~(size_t)255;
    return r;
  };
  _Float16* w1f   = (_Float16*)alloc((size_t)32 * 256 * 2);
  _Float16* w2f   = (_Float16*)alloc((size_t)64 * 512 * 2);
  _Float16* w3f   = (_Float16*)alloc((size_t)64 * 576 * 2);
  _Float16* fc16  = (_Float16*)alloc((size_t)512 * 1024 * 2);
  _Float16* wi16  = (_Float16*)alloc((size_t)1024 * 512 * 2);
  _Float16* wh16  = (_Float16*)alloc((size_t)1024 * 256 * 2);
  _Float16* head16= (_Float16*)alloc((size_t)32 * 256 * 2);
  float*    headB = (float*)   alloc((size_t)32 * 4);
  float*    gates = (float*)   alloc((size_t)32 * 1024 * 4);
  _Float16* c1o   = (_Float16*)alloc((size_t)4096 * 7680 * 2);
  _Float16* c2o   = (_Float16*)alloc((size_t)4096 * 2304 * 2);
  _Float16* c3o   = (_Float16*)alloc((size_t)4096 * 1024 * 2);
  _Float16* feats = (_Float16*)alloc((size_t)4096 * 512 * 2);
  _Float16* outs  = (_Float16*)alloc((size_t)4096 * 256 * 2);

  auto cdiv = [](int a, int b) { return (a + b - 1) / b; };
  cast_f16_kernel<<<cdiv(1024, 256), 256, 0, stream>>>(w1f, w1, 1024);
  cast_f16_kernel<<<cdiv(4096, 256), 256, 0, stream>>>(w2f, w2, 4096);
  cast_f16_kernel<<<cdiv(4608, 256), 256, 0, stream>>>(w3f, w3, 4608);
  cast_f16_kernel<<<cdiv(65536, 256), 256, 0, stream>>>(fc16, fw, 65536);
  cast_f16_kernel<<<cdiv(65536, 256), 256, 0, stream>>>(wi16, wi, 65536);
  cast_f16_kernel<<<cdiv(32768, 256), 256, 0, stream>>>(wh16, wh, 32768);
  head_build_kernel<<<cdiv(32 * 256, 256), 256, 0, stream>>>(head16, headB,
                                                             aw, ab, cw, cb);

  conv1_kernel<<<4096, 256, 0, stream>>>(x, w1f, b1, c1o);
  conv2_kernel<<<4096, 256, 0, stream>>>(c1o, w2f, b2, c2o);
  conv3_kernel<<<4096, 128, 0, stream>>>(c2o, w3f, b3, c3o);
  // fc: (4096x1024) @ (512x1024)^T + relu -> feats f16
  gemm_kernel<<<dim3(64, 8), 256, 0, stream>>>(c3o, fc16, fb, feats, nullptr,
                                               4096, 512, 1024, 512, 512, 1);
  lstm_kernel<<<1, 1024, 0, stream>>>(feats, wi16, wh16, bi, bh, done, h0, c0,
                                      gates, outs);
  // heads: (4096x256) @ (32x256)^T -> d_out f32 [4096][19]
  gemm_kernel<<<dim3(64, 1), 256, 0, stream>>>(outs, head16, headB, nullptr,
                                               (float*)d_out, 4096, 32, 256,
                                               19, 19, 0);
}
